// OPAM_Module_39883066310732
// MI455X (gfx1250) — matrix operand order repaired and checked by reading, not compiled, not run on hardware
//
#include <hip/hip_runtime.h>

typedef __attribute__((ext_vector_type(16))) __bf16 v16bf;
typedef __attribute__((ext_vector_type(8)))  __bf16 v8bf;
typedef __attribute__((ext_vector_type(8)))  float  v8f;

#define DEVINL __device__ __forceinline__

// pack two f32 -> two bf16 in one dword (compiler fuses to v_cvt_pk_bf16_f32)
DEVINL unsigned int pkbf(float a, float b) {
    union { __bf16 h[2]; unsigned int u; } cv;
    cv.h[0] = (__bf16)a; cv.h[1] = (__bf16)b;
    return cv.u;
}

DEVINL v16bf frag_from_lds_a(const __bf16* base /* &tile[m][0] */, int lane) {
    const int ah = (lane >> 4) * 8;
    const v8bf* p0 = (const v8bf*)(base + ah);
    const v8bf* p1 = (const v8bf*)(base + 16 + ah);
    v16bf f;
    #pragma unroll
    for (int i = 0; i < 8; ++i) { f[i] = (*p0)[i]; f[8 + i] = (*p1)[i]; }
    return f;
}

DEVINL v16bf frag_from_lds_b(const __bf16* base /* &tile[n][0] */, int lane) {
    const int bh = (lane >> 4) * 8;
    const v8bf* p0 = (const v8bf*)(base + bh);
    const v8bf* p1 = (const v8bf*)(base + 16 + bh);
    v16bf f;
    #pragma unroll
    for (int i = 0; i < 8; ++i) { f[i] = (*p0)[i]; f[8 + i] = (*p1)[i]; }
    return f;
}

// stage one 16-float K-run (4x b128 loads) as packed bf16 (2x b128 LDS stores)
DEVINL void stage_row32(const float* __restrict__ src, __bf16* dst16 /*16 elems*/) {
    const float4* p = (const float4*)src;
    float4 a0 = p[0], a1 = p[1], a2 = p[2], a3 = p[3];
    uint4 u0, u1;
    u0.x = pkbf(a0.x, a0.y); u0.y = pkbf(a0.z, a0.w);
    u0.z = pkbf(a1.x, a1.y); u0.w = pkbf(a1.z, a1.w);
    u1.x = pkbf(a2.x, a2.y); u1.y = pkbf(a2.z, a2.w);
    u1.z = pkbf(a3.x, a3.y); u1.w = pkbf(a3.z, a3.w);
    uint4* d = (uint4*)dst16;
    d[0] = u0; d[1] = u1;
}

// ---------------------------------------------------------------------------
// Weight-shared channel GEMM (1x1 conv): one block per batch computes
//   Out[b] (256 x L) = W (256 x K) @ X[b] (K x L) + bias
// Block = 512 threads (16 waves). Wave w owns output rows [16w, 16w+16),
// sweeping 10 N-fragments (L=150 padded to 160). Per 32-K step the block
// stages the W slice (256x32) and transposed X slice (160x32) in LDS once.
// ---------------------------------------------------------------------------
__global__ void convgemm_wmma(const float* __restrict__ W,     // [256][K], K-major
                              const float* __restrict__ X, long long sXb, // [K][L]
                              const float* __restrict__ bias,  // [256]
                              float* __restrict__ Out, long long sOb,     // [256][L]
                              int K, int L)
{
    __shared__ __align__(16) __bf16 sW[256 * 32];   // [m][k]
    __shared__ __align__(16) __bf16 sX[160 * 32];   // [l][k] (transposed)

    const int t    = threadIdx.x;
    const int lane = t & 31;
    const int w    = t >> 5;                  // wave index 0..15 == M-fragment
    const int b    = blockIdx.x;
    const float* Xb = X + (long long)b * sXb;

    v8f acc[10];
    #pragma unroll
    for (int nf = 0; nf < 10; ++nf)
        acc[nf] = v8f{0.f, 0.f, 0.f, 0.f, 0.f, 0.f, 0.f, 0.f};

    const int wr = t >> 1;                 // W row 0..255
    const int wc = (t & 1) * 16;           // W k-half (0 or 16)
    const int xl = t >> 1;                 // X l-row 0..159 (t < 320 only)
    const int xh = (t & 1) * 16;           // X k-half
    const int ml = lane & 15;

    for (int kb = 0; kb < K; kb += 32) {
        // ---- stage W slice (256 x 32) ----
        stage_row32(W + (size_t)wr * K + kb + wc, &sW[wr * 32 + wc]);
        // ---- stage X slice transposed (160 x 32): k-contiguous per thread ----
        if (t < 320) {
            const bool lOk = xl < L;
            const float* p = Xb + (size_t)(kb + xh) * L + (lOk ? xl : 0);
            float v[16];
            #pragma unroll
            for (int i = 0; i < 16; ++i) {
                float f = p[i * L];                    // strided global b32
                v[i] = lOk ? f : 0.0f;
            }
            uint4 u0, u1;
            u0.x = pkbf(v[0],  v[1]);  u0.y = pkbf(v[2],  v[3]);
            u0.z = pkbf(v[4],  v[5]);  u0.w = pkbf(v[6],  v[7]);
            u1.x = pkbf(v[8],  v[9]);  u1.y = pkbf(v[10], v[11]);
            u1.z = pkbf(v[12], v[13]); u1.w = pkbf(v[14], v[15]);
            uint4* d = (uint4*)&sX[xl * 32 + xh];
            d[0] = u0; d[1] = u1;
        }
        __syncthreads();

        const v16bf af = frag_from_lds_a(&sW[(w * 16 + ml) * 32], lane);
        #pragma unroll
        for (int nf = 0; nf < 10; ++nf) {
            const v16bf bfv = frag_from_lds_b(&sX[(nf * 16 + ml) * 32], lane);
            acc[nf] = __builtin_amdgcn_wmma_f32_16x16x32_bf16(
                false, af, false, bfv, (short)0, acc[nf], false, false);
        }
        __syncthreads();
    }

    const int mh = 8 * (lane >> 4);
    #pragma unroll
    for (int nf = 0; nf < 10; ++nf) {
        const int n = nf * 16 + ml;
        if (n < L) {
            #pragma unroll
            for (int r = 0; r < 8; ++r) {
                const int m = w * 16 + r + mh;
                Out[(long long)b * sOb + (long long)m * L + n] = acc[nf][r] + bias[m];
            }
        }
    }
}

// ---------------------------------------------------------------------------
// Generic batched WMMA GEMM (one wave = one 16x16 tile), arbitrary strides.
// Branchless staging, full/ragged K-tile split.
// Used for the small attention GEMMs (energy, att-output).
// ---------------------------------------------------------------------------
__global__ void gemm_tile_wmma(const float* __restrict__ A, long long sAb, int sAm, int sAk,
                               const float* __restrict__ B, long long sBb, int sBk, int sBn,
                               float* __restrict__ C, long long sCb, int sCm,
                               const float* __restrict__ resid, long long sRb, int sRm,
                               const float* __restrict__ gammaPtr,
                               int M, int N, int K)
{
    __shared__ __align__(16) __bf16 lA[16 * 32];   // [m][k]
    __shared__ __align__(16) __bf16 lB[16 * 32];   // [n][k] (transposed)

    const int lane = threadIdx.x & 31;
    const int m0 = blockIdx.x * 16;
    const int n0 = blockIdx.y * 16;
    const int b  = blockIdx.z;

    const float* Ab = A + (long long)b * sAb;
    const float* Bb = B + (long long)b * sBb;

    v8f acc = {0.f, 0.f, 0.f, 0.f, 0.f, 0.f, 0.f, 0.f};

    const int rowS = lane >> 1;
    const int colS = (lane & 1) * 16;
    const int gm = m0 + rowS;
    const int gn = n0 + rowS;
    const bool mOk = gm < M;
    const bool nOk = gn < N;
    const float* pA = Ab + (long long)(mOk ? gm : 0) * sAm;   // clamped row base
    const float* pB = Bb + (long long)(nOk ? gn : 0) * sBn;

    for (int kb = 0; kb < K; kb += 32) {
        float va[16], vb[16];
        if (kb + 32 <= K) {              // full K tile: no k masking
            const int k0 = kb + colS;
            #pragma unroll
            for (int i = 0; i < 16; ++i) {
                float a = pA[(k0 + i) * sAk];
                float bv_ = pB[(k0 + i) * sBk];
                va[i] = mOk ? a : 0.0f;
                vb[i] = nOk ? bv_ : 0.0f;
            }
        } else {                         // ragged tail
            #pragma unroll
            for (int i = 0; i < 16; ++i) {
                const int gk = kb + colS + i;
                const int ko = (gk < K) ? gk : 0;
                float a = pA[ko * sAk];
                float bv_ = pB[ko * sBk];
                va[i] = (mOk && gk < K) ? a : 0.0f;
                vb[i] = (nOk && gk < K) ? bv_ : 0.0f;
            }
        }
        {
            uint4 ua0, ua1, ub0, ub1;
            ua0.x = pkbf(va[0],  va[1]);  ua0.y = pkbf(va[2],  va[3]);
            ua0.z = pkbf(va[4],  va[5]);  ua0.w = pkbf(va[6],  va[7]);
            ua1.x = pkbf(va[8],  va[9]);  ua1.y = pkbf(va[10], va[11]);
            ua1.z = pkbf(va[12], va[13]); ua1.w = pkbf(va[14], va[15]);
            ub0.x = pkbf(vb[0],  vb[1]);  ub0.y = pkbf(vb[2],  vb[3]);
            ub0.z = pkbf(vb[4],  vb[5]);  ub0.w = pkbf(vb[6],  vb[7]);
            ub1.x = pkbf(vb[8],  vb[9]);  ub1.y = pkbf(vb[10], vb[11]);
            ub1.z = pkbf(vb[12], vb[13]); ub1.w = pkbf(vb[14], vb[15]);
            uint4* dA = (uint4*)&lA[rowS * 32 + colS];
            uint4* dB = (uint4*)&lB[rowS * 32 + colS];
            dA[0] = ua0; dA[1] = ua1;
            dB[0] = ub0; dB[1] = ub1;
        }
        __syncthreads();

        const int ml = lane & 15;
        const v16bf af  = frag_from_lds_a(&lA[ml * 32], lane);
        const v16bf bfv = frag_from_lds_b(&lB[ml * 32], lane);
        acc = __builtin_amdgcn_wmma_f32_16x16x32_bf16(
            false, af, false, bfv, (short)0, acc, false, false);
        __syncthreads();
    }

    const int n = n0 + (lane & 15);
    if (n < N) {
        const float g = resid ? gammaPtr[0] : 0.0f;
        #pragma unroll
        for (int r = 0; r < 8; ++r) {
            const int m = m0 + r + 8 * (lane >> 4);
            if (m < M) {
                float v = acc[r];
                if (resid)
                    v = g * v + resid[(long long)b * sRb + (long long)m * sRm + n];
                C[(long long)b * sCb + (long long)m * sCm + n] = v;
            }
        }
    }
}

// ---------------------------------------------------------------------------
// Row softmax over L=150 elements, one wave (32 lanes) per row, in place.
// ---------------------------------------------------------------------------
__global__ void softmax_rows(float* __restrict__ E, int L)
{
    const int lane = threadIdx.x;
    float* row = E + (long long)blockIdx.x * L;

    float vals[5];
    float mx = -3.4e38f;
    #pragma unroll
    for (int i = 0; i < 5; ++i) {
        const int j = lane + 32 * i;
        const float v = (j < L) ? row[j] : -3.4e38f;
        vals[i] = v;
        mx = fmaxf(mx, v);
    }
    #pragma unroll
    for (int off = 16; off > 0; off >>= 1)
        mx = fmaxf(mx, __shfl_xor(mx, off, 32));

    float sum = 0.0f;
    #pragma unroll
    for (int i = 0; i < 5; ++i) {
        const int j = lane + 32 * i;
        const float e = (j < L) ? __expf(vals[i] - mx) : 0.0f;
        vals[i] = e;
        sum += e;
    }
    #pragma unroll
    for (int off = 16; off > 0; off >>= 1)
        sum += __shfl_xor(sum, off, 32);

    const float inv = 1.0f / sum;
    #pragma unroll
    for (int i = 0; i < 5; ++i) {
        const int j = lane + 32 * i;
        if (j < L) row[j] = vals[i] * inv;
    }
}

// ---------------------------------------------------------------------------
// K-major FC GEMM with LDS tiling: C[m][n] = sum_k A[m][k]*W[n][k] + bias[n]
// Block = 256 threads (8 waves), block tile 128M x 128N, K-step 32.
// Per K-step the block stages A-slice (128x32) and W-slice (128x32) in LDS
// once (removes the 4x/2x intra-block global re-reads), then each wave
// computes 64M x 32N = 4x2 WMMA fragments from LDS.
// Requires M % 128 == 0, N % 128 == 0, K % 32 == 0.
// ---------------------------------------------------------------------------
__global__ void fc_wmma(const float* __restrict__ A, int lda,
                        const float* __restrict__ W, int ldw,
                        const float* __restrict__ bias,
                        float* __restrict__ C, int N, int K)
{
    __shared__ __align__(16) __bf16 sA[128 * 32];   // [m][k]
    __shared__ __align__(16) __bf16 sB[128 * 32];   // [n][k]

    const int t    = threadIdx.x;
    const int lane = t & 31;
    const int widx = t >> 5;
    const int wm   = (widx >> 2) * 64;    // wave M offset in block tile
    const int wn   = (widx & 3) * 32;     // wave N offset in block tile
    const int m0b  = blockIdx.x * 128;
    const int n0b  = blockIdx.y * 128;

    const int sr = t >> 1;                // staging row 0..127
    const int sc = (t & 1) * 16;          // staging k-half
    const int ml = lane & 15;

    v8f acc[4][2];
    #pragma unroll
    for (int i = 0; i < 4; ++i)
        #pragma unroll
        for (int j = 0; j < 2; ++j)
            acc[i][j] = v8f{0.f, 0.f, 0.f, 0.f, 0.f, 0.f, 0.f, 0.f};

    for (int kb = 0; kb < K; kb += 32) {
        const float* pA = A + (size_t)(m0b + sr) * lda + kb + sc;
        const float* pW = W + (size_t)(n0b + sr) * ldw + kb + sc;
        stage_row32(pA, &sA[sr * 32 + sc]);
        stage_row32(pW, &sB[sr * 32 + sc]);
        if (kb + 32 < K)
            __builtin_prefetch(pW + 32, 0, 1);   // global_prefetch_b8, next K chunk
        __syncthreads();

        v16bf af[4], bfv[2];
        #pragma unroll
        for (int i = 0; i < 4; ++i)
            af[i] = frag_from_lds_a(&sA[(wm + 16 * i + ml) * 32], lane);
        #pragma unroll
        for (int j = 0; j < 2; ++j)
            bfv[j] = frag_from_lds_b(&sB[(wn + 16 * j + ml) * 32], lane);

        #pragma unroll
        for (int i = 0; i < 4; ++i)
            #pragma unroll
            for (int j = 0; j < 2; ++j)
                acc[i][j] = __builtin_amdgcn_wmma_f32_16x16x32_bf16(
                    false, af[i], false, bfv[j], (short)0, acc[i][j], false, false);
        __syncthreads();
    }

    const int mh = 8 * (lane >> 4);
    #pragma unroll
    for (int i = 0; i < 4; ++i)
        #pragma unroll
        for (int j = 0; j < 2; ++j) {
            const int n = n0b + wn + 16 * j + ml;
            const float bn = bias[n];
            #pragma unroll
            for (int r = 0; r < 8; ++r) {
                const int m = m0b + wm + 16 * i + r + mh;
                C[(size_t)m * N + n] = acc[i][j][r] + bn;
            }
        }
}

// ---------------------------------------------------------------------------

extern "C" void kernel_launch(void* const* d_in, const int* in_sizes, int n_in,
                              void* d_out, int out_size, void* d_ws, size_t ws_size,
                              hipStream_t stream)
{
    (void)in_sizes; (void)n_in; (void)out_size; (void)ws_size;

    const float* x     = (const float*)d_in[0];
    const float* Wv    = (const float*)d_in[1];
    const float* bv    = (const float*)d_in[2];
    const float* Wq    = (const float*)d_in[3];
    const float* bq    = (const float*)d_in[4];
    const float* Wk    = (const float*)d_in[5];
    const float* bk    = (const float*)d_in[6];
    const float* gamma = (const float*)d_in[7];
    const float* W1    = (const float*)d_in[8];
    const float* b1    = (const float*)d_in[9];
    const float* W2    = (const float*)d_in[10];
    const float* b2    = (const float*)d_in[11];
    float* out = (float*)d_out;
    float* ws  = (float*)d_ws;

    const int B = 256, CIN = 2048, L = 150, C = 256, H1 = 8192, OUTF = 2048;
    const long long CL = (long long)C * L;    // 38400
    const long long LL = (long long)L * L;    // 22500

    float* pv  = ws;                          // [B][C][L]
    float* qb  = pv  + (size_t)B * CL;        // [B][C][L]
    float* kb_ = qb  + (size_t)B * CL;        // [B][C][L]
    float* att = kb_ + (size_t)B * CL;        // [B][L][L]
    float* ob  = att + (size_t)B * LL;        // [B][C][L]  (== flat [B][C*L])
    float* hb  = ob  + (size_t)B * CL;        // [B][H1]

    // pv = Wv @ x + bv         (per batch, W read once per batch)
    convgemm_wmma<<<dim3(B), dim3(512), 0, stream>>>(
        Wv, x, (long long)CIN * L, bv, pv, CL, CIN, L);

    // q = Wq @ pv + bq
    convgemm_wmma<<<dim3(B), dim3(512), 0, stream>>>(
        Wq, pv, CL, bq, qb, CL, C, L);

    // k = Wk @ pv + bk
    convgemm_wmma<<<dim3(B), dim3(512), 0, stream>>>(
        Wk, pv, CL, bk, kb_, CL, C, L);

    // energy[i][j] = sum_c q[c][i] * k[c][j]   (M=L, N=L, K=C)
    gemm_tile_wmma<<<dim3(10, 10, B), dim3(32), 0, stream>>>(
        qb, CL, 1, L,          // A[m=i][k=c] = q[c*L + i]
        kb_, CL, L, 1,         // B[k=c][n=j] = k[c*L + j]
        att, LL, L,
        nullptr, 0, 0, nullptr,
        L, L, C);

    // att = softmax(energy, axis=-1), in place
    softmax_rows<<<dim3(B * L), dim3(32), 0, stream>>>(att, L);

    // out = gamma * (pv @ att^T) + pv    (M=C, N=L, K=L)
    gemm_tile_wmma<<<dim3(16, 10, B), dim3(32), 0, stream>>>(
        pv, CL, L, 1,          // A[m=c][k=mm] = pv[c*L + mm]
        att, LL, 1, L,         // B[k=mm][n=l] = att[l*L + mm]
        ob, CL, L,
        pv, CL, L, gamma,
        C, L, L);

    // h = flat @ W1^T + b1    (M=256, N=8192, K=38400)
    fc_wmma<<<dim3(2, 64), dim3(256), 0, stream>>>(
        ob, (int)CL, W1, (int)CL, b1, hb, H1, (int)CL);

    // out = h @ W2^T + b2     (M=256, N=2048, K=8192)
    fc_wmma<<<dim3(2, 16), dim3(256), 0, stream>>>(
        hb, H1, W2, H1, b2, out, OUTF, H1);
}
